// ResidualVectorQuantizer_71347996721152
// MI455X (gfx1250) — matrix operand order repaired and checked by reading, not compiled, not run on hardware
//
#include <hip/hip_runtime.h>
#include <hip/hip_bf16.h>
#include <math.h>

#define N_TOKENS 65536
#define CB_SIZE  1024
#define CB_DIM   256
#define NUM_CB   8

typedef __attribute__((ext_vector_type(16))) __bf16 v16bf;
typedef __attribute__((ext_vector_type(8)))  float  v8f;

union FragU { v16bf v; unsigned int u[8]; };

__device__ __forceinline__ unsigned int pack_bf16_rne(float a, float b) {
  unsigned int ua = __float_as_uint(a);
  unsigned int ub = __float_as_uint(b);
  ua += 0x7fffu + ((ua >> 16) & 1u);   // round-to-nearest-even to bf16
  ub += 0x7fffu + ((ub >> 16) & 1u);
  return (ua >> 16) | (ub & 0xffff0000u);
}

// ---------------------------------------------------------------------------
// Init: residual = z ; qsum = 0 ; counts = 0 ; per-layer sumsq = 0
// ---------------------------------------------------------------------------
__global__ void rvq_init(const float* __restrict__ z, float* __restrict__ residual,
                         float* __restrict__ qsum, float* __restrict__ counts,
                         float* __restrict__ sumsq) {
  size_t gid = (size_t)blockIdx.x * blockDim.x + threadIdx.x;
  if (gid < (size_t)N_TOKENS * CB_DIM) {
    residual[gid] = z[gid];
    qsum[gid]     = 0.0f;
  }
  if (gid < NUM_CB * CB_SIZE) counts[gid] = 0.0f;
  if (gid < NUM_CB)           sumsq[gid]  = 0.0f;
}

// ---------------------------------------------------------------------------
// Per layer: e_sq[k] = ||codebook[k]||^2  AND  bf16-convert the codebook row
// (one block per codebook row; conversion amortized over 512 GEMM blocks)
// ---------------------------------------------------------------------------
__global__ void rvq_esq_cvt(const float* __restrict__ cb, float* __restrict__ esq,
                            unsigned int* __restrict__ cb16) {
  __shared__ float red[256];
  const int row = blockIdx.x;
  const float v = cb[(size_t)row * CB_DIM + threadIdx.x];
  red[threadIdx.x] = v * v;
  if (threadIdx.x < 128) {
    const float* src = cb + (size_t)row * CB_DIM + threadIdx.x * 2;
    cb16[row * 128 + threadIdx.x] = pack_bf16_rne(src[0], src[1]);
  }
  __syncthreads();
  for (int s = 128; s > 0; s >>= 1) {
    if (threadIdx.x < s) red[threadIdx.x] += red[threadIdx.x + s];
    __syncthreads();
  }
  if (threadIdx.x == 0) esq[row] = red[0];
}

// ---------------------------------------------------------------------------
// WMMA argmin kernel: per block 128 rows (8 waves x 16 rows), full K=1024.
// score[m][k] = e_sq[k] - 2 * dot(residual[m], code[k])  (z_sq is argmin-invariant)
// Double-buffered LDS staging of the bf16 codebook; 8 back-to-back WMMAs/chunk.
// ---------------------------------------------------------------------------
__global__ void __launch_bounds__(256)
rvq_argmin(const float* __restrict__ residual,
           const uint4* __restrict__ cb16,    // bf16 codebook, 512 uint4 per 16-row chunk
           const float* __restrict__ esq,     // [1024]
           int* __restrict__ idx_out) {
  __shared__ uint4 ldsB[2][512];              // 2 x 8KB double buffer
  __shared__ float ldsEsq[CB_SIZE];           // 4KB

  const int tid  = threadIdx.x;
  const int lane = tid & 31;
  const int wave = tid >> 5;
  const int col  = lane & 15;
  const int half = lane >> 4;
  const int rowBase = blockIdx.x * 128 + wave * 16;

  // Stage e_sq once.
  for (int j = tid; j < CB_SIZE; j += 256) ldsEsq[j] = esq[j];

  // Build resident A-fragments (16x32 bf16) for all 8 D-chunks, converted on the fly.
  // A layout (16-bit, 16x32): VGPR i<4 -> K = half*8 + 2i ; i>=4 -> K = 16 + half*8 + 2(i-4)
  FragU afrag[8];
  const float* rrow = residual + (size_t)(rowBase + col) * CB_DIM;
#pragma unroll
  for (int c = 0; c < 8; ++c) {
#pragma unroll
    for (int i = 0; i < 8; ++i) {
      int k = (i < 4) ? (half * 8 + 2 * i) : (16 + half * 8 + 2 * (i - 4));
      int d = c * 32 + k;
      afrag[c].u[i] = pack_bf16_rne(rrow[d], rrow[d + 1]);
    }
  }

  // Prologue: stage chunk 0 (b128 copies, already bf16).
  ldsB[0][tid]       = cb16[tid];
  ldsB[0][tid + 256] = cb16[tid + 256];
  __syncthreads();

  float best[8];
  int   bidx[8];
#pragma unroll
  for (int j = 0; j < 8; ++j) { best[j] = 3.4e38f; bidx[j] = 0; }

  for (int kc = 0; kc < CB_SIZE / 16; ++kc) {
    const int cur = kc & 1;

    // Issue global loads for the NEXT chunk early so they overlap the WMMAs.
    uint4 n0, n1;
    const bool hasNext = (kc + 1) < (CB_SIZE / 16);
    if (hasNext) {
      n0 = cb16[(size_t)(kc + 1) * 512 + tid];
      n1 = cb16[(size_t)(kc + 1) * 512 + 256 + tid];
      if (kc + 2 < (CB_SIZE / 16))
        __builtin_prefetch(&cb16[(size_t)(kc + 2) * 512 + tid], 0, 0);
    }

    // Bulk-load all 8 B-fragments for this chunk (16x ds_load_b128, one wait),
    // then fire 8 WMMAs back-to-back.
    // B layout (32x16 bf16): VGPR i<4 -> K = half*8 + {2i, 2i+1};
    // i>=4 -> K = 16 + half*8 + {2(i-4), 2(i-4)+1}; N = col (codebook row within chunk).
    FragU bfr[8];
    const uint4* bu4 = &ldsB[cur][(col << 5) + half];
#pragma unroll
    for (int c = 0; c < 8; ++c) {
      *(uint4*)&bfr[c].u[0] = bu4[c * 4];
      *(uint4*)&bfr[c].u[4] = bu4[c * 4 + 2];
    }

    v8f acc = {};
#pragma unroll
    for (int c = 0; c < 8; ++c) {
      acc = __builtin_amdgcn_wmma_f32_16x16x32_bf16(
          false, afrag[c].v, false, bfr[c].v, (short)0, acc, false, false);
    }

    const int nglob = kc * 16 + col;
    const float es = ldsEsq[nglob];
#pragma unroll
    for (int j = 0; j < 8; ++j) {
      float s = fmaf(-2.0f, acc[j], es);
      if (s < best[j]) { best[j] = s; bidx[j] = nglob; }   // strict < keeps first index
    }

    // Store the prefetched next chunk into the alternate buffer.
    if (hasNext) {
      ldsB[cur ^ 1][tid]       = n0;
      ldsB[cur ^ 1][tid + 256] = n1;
    }
    __syncthreads();   // single barrier per iteration (double buffering)
  }

  // Min+index reduction across the 16 lanes of each half (C layout:
  // lanes 0-15 hold rows 0-7, lanes 16-31 hold rows 8-15; VGPR j -> row offset j).
#pragma unroll
  for (int j = 0; j < 8; ++j) {
#pragma unroll
    for (int off = 8; off >= 1; off >>= 1) {
      float ov = __shfl_xor(best[j], off, 32);
      int   oi = __shfl_xor(bidx[j], off, 32);
      if (ov < best[j] || (ov == best[j] && oi < bidx[j])) { best[j] = ov; bidx[j] = oi; }
    }
  }
  if (col == 0) {
#pragma unroll
    for (int j = 0; j < 8; ++j) {
      int row = rowBase + half * 8 + j;
      idx_out[row] = bidx[j];
    }
  }
}

// ---------------------------------------------------------------------------
// Per-token update: gather code row, residual -= q, qsum += q, loss + counts.
// One wave per token (8 tokens per 256-thread block).
// ---------------------------------------------------------------------------
__global__ void __launch_bounds__(256)
rvq_update(float* __restrict__ residual, float* __restrict__ qsum,
           const float* __restrict__ cb,
           const int* __restrict__ idx_in,
           float* __restrict__ counts,        // this layer's 1024 counters
           float* __restrict__ sumsq,         // this layer's loss accumulator
           float* __restrict__ idx_f_out) {   // d_out index slice for this layer
  const int lane  = threadIdx.x & 31;
  const int wave  = threadIdx.x >> 5;
  const int token = blockIdx.x * 8 + wave;
  const int idx   = idx_in[token];
  const float* crow = cb + (size_t)idx * CB_DIM;
  float* rrow = residual + (size_t)token * CB_DIM;
  float* qrow = qsum     + (size_t)token * CB_DIM;
  float ss = 0.0f;
#pragma unroll
  for (int i = 0; i < 8; ++i) {
    int d = lane + 32 * i;                    // coalesced across the wave
    float q  = crow[d];
    float nr = rrow[d] - q;                   // new residual == (old residual - q)
    rrow[d] = nr;
    qrow[d] += q;
    ss += nr * nr;                            // == (residual - quantized)^2
  }
#pragma unroll
  for (int off = 16; off >= 1; off >>= 1) ss += __shfl_xor(ss, off, 32);
  if (lane == 0) {
    atomicAdd(sumsq, ss);
    atomicAdd(&counts[idx], 1.0f);
    idx_f_out[token] = (float)idx;
  }
}

// ---------------------------------------------------------------------------
// Finalize: perplexity per layer from counts, loss scalars.
// ---------------------------------------------------------------------------
__global__ void rvq_finalize(const float* __restrict__ counts,
                             const float* __restrict__ sumsq,
                             float* __restrict__ out_scalars) {
  __shared__ float red[256];
  __shared__ float perp_acc;
  const int tid = threadIdx.x;
  if (tid == 0) perp_acc = 0.0f;
  __syncthreads();
  for (int l = 0; l < NUM_CB; ++l) {
    float part = 0.0f;
    for (int k = tid; k < CB_SIZE; k += 256) {
      float p = counts[l * CB_SIZE + k] * (1.0f / (float)N_TOKENS);
      part += p * logf(p + 1e-10f);
    }
    red[tid] = part;
    __syncthreads();
    for (int s = 128; s > 0; s >>= 1) {
      if (tid < s) red[tid] += red[tid + s];
      __syncthreads();
    }
    if (tid == 0) perp_acc += expf(-red[0]);
    __syncthreads();
  }
  if (tid == 0) {
    float tc = 0.0f;
    for (int l = 0; l < NUM_CB; ++l) tc += sumsq[l];
    tc *= 1.0f / ((float)N_TOKENS * (float)CB_DIM);   // sum of per-layer means
    out_scalars[0] = 1.25f * tc;                      // vq_loss = 0.25*tc + tcb (tcb==tc)
    out_scalars[1] = tc;                              // total_commit
    out_scalars[2] = tc;                              // total_cb
    out_scalars[3] = perp_acc * (1.0f / NUM_CB);      // avg_perplexity
  }
}

// ---------------------------------------------------------------------------
extern "C" void kernel_launch(void* const* d_in, const int* in_sizes, int n_in,
                              void* d_out, int out_size, void* d_ws, size_t ws_size,
                              hipStream_t stream) {
  const float* z         = (const float*)d_in[0];   // [65536, 256]
  const float* codebooks = (const float*)d_in[1];   // [8, 1024, 256]
  float* out = (float*)d_out;

  float* qsum    = out;                                   // N*D
  float* idx_f   = out + (size_t)N_TOKENS * CB_DIM;       // 8*N (indices as float)
  float* scalars = idx_f + (size_t)NUM_CB * N_TOKENS;     // 4 scalars

  char* ws = (char*)d_ws;
  float*        residual = (float*)ws;        ws += (size_t)N_TOKENS * CB_DIM * sizeof(float);
  float*        esq      = (float*)ws;        ws += (size_t)CB_SIZE * sizeof(float);
  float*        counts   = (float*)ws;        ws += (size_t)NUM_CB * CB_SIZE * sizeof(float);
  float*        sumsq    = (float*)ws;        ws += 64;
  unsigned int* cb16     = (unsigned int*)ws; ws += (size_t)CB_SIZE * (CB_DIM / 2) * sizeof(unsigned int);
  int*          idx_buf  = (int*)ws;

  rvq_init<<<(N_TOKENS * CB_DIM) / 256, 256, 0, stream>>>(z, residual, qsum, counts, sumsq);

  for (int l = 0; l < NUM_CB; ++l) {
    const float* cbl = codebooks + (size_t)l * CB_SIZE * CB_DIM;
    rvq_esq_cvt<<<CB_SIZE, 256, 0, stream>>>(cbl, esq, cb16);
    rvq_argmin<<<N_TOKENS / 128, 256, 0, stream>>>(residual, (const uint4*)cb16, esq, idx_buf);
    rvq_update<<<N_TOKENS / 8, 256, 0, stream>>>(residual, qsum, cbl, idx_buf,
                                                 counts + (size_t)l * CB_SIZE,
                                                 sumsq + l,
                                                 idx_f + (size_t)l * N_TOKENS);
  }

  rvq_finalize<<<1, 256, 0, stream>>>(counts, sumsq, scalars);
}
